// Classifier_57629871177811
// MI455X (gfx1250) — matrix operand order repaired and checked by reading, not compiled, not run on hardware
//
#include <hip/hip_runtime.h>

// ---- CDNA5 (gfx1250) wave32 WMMA types ----
typedef __attribute__((ext_vector_type(16))) __bf16       v16bf;
typedef __attribute__((ext_vector_type(8)))  float        v8f;
typedef __attribute__((ext_vector_type(4)))  unsigned int u32x4;

union FragB16 { u32x4 u[2]; v16bf v; __bf16 h[16]; };

// ---------------------------------------------------------------------------
// Prep: fuse W_i2h / W_i2o into Wcat[512][512] (bf16), convert W_o2o to bf16,
// fuse biases. One-time per launch; negligible vs. the 206-GFLOP scan.
// ---------------------------------------------------------------------------
__global__ void prep_weights(const float* __restrict__ Wi2h,
                             const float* __restrict__ bi2h,
                             const float* __restrict__ Wi2o,
                             const float* __restrict__ bi2o,
                             const float* __restrict__ Wo2o,
                             __bf16* __restrict__ wcat,
                             __bf16* __restrict__ wo2o,
                             float*  __restrict__ bcat) {
  int idx = blockIdx.x * 256 + threadIdx.x;
  if (idx < 512 * 512) {
    int n = idx >> 9, k = idx & 511;
    float v = (n < 256) ? Wi2h[n * 512 + k] : Wi2o[(n - 256) * 512 + k];
    wcat[idx] = (__bf16)v;
  } else if (idx < 512 * 512 + 256 * 512) {
    int j = idx - 512 * 512;
    wo2o[j] = (__bf16)Wo2o[j];
  } else if (idx < 512 * 512 + 256 * 512 + 512) {
    int b = idx - (512 * 512 + 256 * 512);
    bcat[b] = (b < 256) ? bi2h[b] : bi2o[b - 256];
  }
}

// ---------------------------------------------------------------------------
// Persistent RNN scan. 16 workgroups x 256 threads (8 wave32).
// Each workgroup owns 16 batch rows for all T=1024 steps.
//   icbuf[16][512] = [e | h]   (bf16, LDS)
//   ocbuf[16][512] = [h_new|o] (bf16, LDS)
// GEMM1: wave w -> N-tiles 4w..4w+3 of oc (4 wmma per 32-K slab)
// GEMM2: wave w -> N-tiles 2w..2w+1 of out; C regs accumulate the time-sum.
// K loops capped at unroll 2: keeps ~130 VGPRs live -> no scratch spills.
// ---------------------------------------------------------------------------
__global__ __launch_bounds__(256, 1)
void rnn_scan_wmma(const int*    __restrict__ x,
                   const float*  __restrict__ emb,
                   const __bf16* __restrict__ wcat,
                   const __bf16* __restrict__ wo2o,
                   const float*  __restrict__ bcat,
                   const float*  __restrict__ b_o2o,
                   const float*  __restrict__ W_pred,
                   const float*  __restrict__ b_pred,
                   float*        __restrict__ out) {
  __shared__ __bf16 icbuf[16][512];
  __shared__ __bf16 ocbuf[16][512];
  __shared__ float  encs[16][256];
  __shared__ float  logs[16][64];

  const int tid  = threadIdx.x;
  const int lane = tid & 31;
  const int wv   = tid >> 5;        // wave 0..7
  const int wg   = blockIdx.x;      // batch tile 0..15

  // per-lane WMMA fragment coordinates (ISA 7.12.2, wave32)
  const int mrow = lane & 15;       // A row / B column within 16x16 tile
  const int half = lane >> 4;       // lane group
  const int kA   = half * 16;       // A: K run    {kA..kA+15}
  const int kB   = half * 16;       // B: K run    {kB..kB+15}

  // h0 = 0
  {
    u32x4 z = (u32x4)0u;
    for (int i = tid; i < 512; i += 256) {
      int r = i >> 5, c = i & 31;
      ((u32x4*)&icbuf[r][256])[c] = z;
    }
  }

  v8f sum0 = (v8f)0.f, sum1 = (v8f)0.f;   // time-sum of out tiles (C regs)

  for (int t = 0; t < 1024; ++t) {
    // ---- embedding gather: e -> icbuf[:,0:256] (f32 -> bf16) ----
    {
      int r  = tid >> 4;
      int c0 = (tid & 15) << 4;
      int tok = x[t * 256 + wg * 16 + r];
      const float* er = emb + (size_t)tok * 256 + c0;
      FragB16 p;
#pragma unroll
      for (int i = 0; i < 16; ++i) p.h[i] = (__bf16)er[i];
      *(u32x4*)&icbuf[r][c0]     = p.u[0];
      *(u32x4*)&icbuf[r][c0 + 8] = p.u[1];
    }
    __syncthreads();

    // ---- GEMM1: oc[16,512] = ic[16,512] @ Wcat^T ----
    v8f acc[4];
    acc[0] = acc[1] = acc[2] = acc[3] = (v8f)0.f;
#pragma unroll 2
    for (int kk = 0; kk < 512; kk += 32) {
      FragB16 a;                                          // shared across 4 N-tiles
      a.u[0] = *(const u32x4*)&icbuf[mrow][kk + kA];
      a.u[1] = *(const u32x4*)&icbuf[mrow][kk + kA + 8];
#pragma unroll
      for (int nt = 0; nt < 4; ++nt) {
        const __bf16* wr = wcat + (size_t)((wv * 4 + nt) * 16 + mrow) * 512 + kk + kB;
        FragB16 b;
        b.u[0] = *(const u32x4*)wr;
        b.u[1] = *(const u32x4*)(wr + 8);
        acc[nt] = __builtin_amdgcn_wmma_f32_16x16x32_bf16(
            false, a.v, false, b.v, (short)0, acc[nt], false, false);
      }
    }
    // bias + repack oc into LDS as bf16 (C layout: VGPR i -> M=half*8+i, N fixed)
#pragma unroll
    for (int nt = 0; nt < 4; ++nt) {
      int n = (wv * 4 + nt) * 16 + mrow;
      float bn = bcat[n];
#pragma unroll
      for (int i = 0; i < 8; ++i)
        ocbuf[half * 8 + i][n] = (__bf16)(acc[nt][i] + bn);
    }
    __syncthreads();

    // ---- carry h_new: ocbuf[:,0:256] -> icbuf[:,256:512] ----
    // issued first so the ds stores drain underneath the GEMM2 wmma stream
    {
      int base = tid * 2;
#pragma unroll
      for (int rep = 0; rep < 2; ++rep) {
        int i = base + rep;
        int r = i >> 5, c = i & 31;
        ((u32x4*)&icbuf[r][256])[c] = ((const u32x4*)&ocbuf[r][0])[c];
      }
    }

    // ---- GEMM2: out[16,256] += oc[16,512] @ Wo2o^T (time-sum in C regs) ----
#pragma unroll 2
    for (int kk = 0; kk < 512; kk += 32) {
      FragB16 a;
      a.u[0] = *(const u32x4*)&ocbuf[mrow][kk + kA];
      a.u[1] = *(const u32x4*)&ocbuf[mrow][kk + kA + 8];
      {
        const __bf16* wr = wo2o + (size_t)((wv * 2 + 0) * 16 + mrow) * 512 + kk + kB;
        FragB16 b;
        b.u[0] = *(const u32x4*)wr;
        b.u[1] = *(const u32x4*)(wr + 8);
        sum0 = __builtin_amdgcn_wmma_f32_16x16x32_bf16(
            false, a.v, false, b.v, (short)0, sum0, false, false);
      }
      {
        const __bf16* wr = wo2o + (size_t)((wv * 2 + 1) * 16 + mrow) * 512 + kk + kB;
        FragB16 b;
        b.u[0] = *(const u32x4*)wr;
        b.u[1] = *(const u32x4*)(wr + 8);
        sum1 = __builtin_amdgcn_wmma_f32_16x16x32_bf16(
            false, a.v, false, b.v, (short)0, sum1, false, false);
      }
    }
    __syncthreads();
  }

  // ---- encoded = time-sum / T + b_o2o ----
#pragma unroll
  for (int j = 0; j < 2; ++j) {
    v8f s = j ? sum1 : sum0;
    int n = (wv * 2 + j) * 16 + mrow;
    float bo = b_o2o[n];
#pragma unroll
    for (int i = 0; i < 8; ++i)
      encs[half * 8 + i][n] = s[i] * (1.0f / 1024.0f) + bo;
  }
  __syncthreads();

  // ---- logits[16,64] = encoded @ W_pred^T + b_pred (tiny; VALU) ----
  {
    int r  = tid >> 4;
    int o0 = (tid & 15) << 2;
    float a0 = 0.f, a1 = 0.f, a2 = 0.f, a3 = 0.f;
    for (int k = 0; k < 256; ++k) {
      float ev = encs[r][k];
      a0 += ev * W_pred[(o0 + 0) * 256 + k];
      a1 += ev * W_pred[(o0 + 1) * 256 + k];
      a2 += ev * W_pred[(o0 + 2) * 256 + k];
      a3 += ev * W_pred[(o0 + 3) * 256 + k];
    }
    logs[r][o0 + 0] = a0 + b_pred[o0 + 0];
    logs[r][o0 + 1] = a1 + b_pred[o0 + 1];
    logs[r][o0 + 2] = a2 + b_pred[o0 + 2];
    logs[r][o0 + 3] = a3 + b_pred[o0 + 3];
  }
  __syncthreads();

  // ---- log-softmax / argmax / probs per batch row ----
  if (tid < 16) {
    int r = tid, gb = wg * 16 + r;
    float mx = logs[r][0];
    int am = 0;
    for (int o = 1; o < 64; ++o) {
      float v = logs[r][o];
      if (v > mx) { mx = v; am = o; }
    }
    float se = 0.f;
    for (int o = 0; o < 64; ++o) se += __expf(logs[r][o] - mx);
    float lse = __logf(se);
    float* lp = out + 256 + gb * 64;
    float* pp = out + 256 + 16384 + gb * 64;
    for (int o = 0; o < 64; ++o) {
      float l = logs[r][o] - mx - lse;
      lp[o] = l;
      pp[o] = __expf(l);
    }
    out[gb] = (float)am;   // preds
  }
}

// ---------------------------------------------------------------------------
extern "C" void kernel_launch(void* const* d_in, const int* in_sizes, int n_in,
                              void* d_out, int out_size, void* d_ws, size_t ws_size,
                              hipStream_t stream) {
  const int*   x     = (const int*)  d_in[0];
  const float* emb   = (const float*)d_in[1];
  const float* Wi2h  = (const float*)d_in[2];
  const float* bi2h  = (const float*)d_in[3];
  const float* Wi2o  = (const float*)d_in[4];
  const float* bi2o  = (const float*)d_in[5];
  const float* Wo2o  = (const float*)d_in[6];
  const float* bo2o  = (const float*)d_in[7];
  const float* Wpred = (const float*)d_in[8];
  const float* bpred = (const float*)d_in[9];

  char* ws = (char*)d_ws;
  __bf16* wcat    = (__bf16*)ws;                                   // 512*512 bf16
  __bf16* wo2o_bf = (__bf16*)(ws + 512 * 512 * 2);                 // 256*512 bf16
  float*  bcat    = (float*) (ws + 512 * 512 * 2 + 256 * 512 * 2); // 512 f32

  int total = 512 * 512 + 256 * 512 + 512;
  prep_weights<<<(total + 255) / 256, 256, 0, stream>>>(
      Wi2h, bi2h, Wi2o, bi2o, Wo2o, wcat, wo2o_bf, bcat);

  rnn_scan_wmma<<<16, 256, 0, stream>>>(
      x, emb, wcat, wo2o_bf, bcat, bo2o, Wpred, bpred, (float*)d_out);
}
